// FastSpeech_88218628260704
// MI455X (gfx1250) — matrix operand order repaired and checked by reading, not compiled, not run on hardware
//
#include <hip/hip_runtime.h>
#include <hip/hip_bf16.h>
#include <math.h>

typedef __bf16 bf16;
typedef __bf16 v16bf __attribute__((ext_vector_type(16)));
typedef float  v8f   __attribute__((ext_vector_type(8)));

#define D_MODEL 384
#define D_FF    1536
#define N_HEADS 2
#define DH      192
#define BS_     16
#define L_PH    128
#define T_MEL   512
#define N_MELS  80

// ---------------- weight cast + transpose: fp32 (K,N) -> bf16 (N,K) ----------------
// grid: (ceil(N/256), K); coalesced reads along n, no integer division.
__global__ void k_castT(const float* __restrict__ in, bf16* __restrict__ out, int K, int N) {
  int n = blockIdx.x * blockDim.x + threadIdx.x;
  int k = blockIdx.y;
  if (n >= N) return;
  out[(long)n * K + k] = (bf16)in[(long)k * N + n];
}

// ---------------- sinusoidal positional encoding ----------------
__device__ __forceinline__ float posenc(int l, int d) {
  float i2 = (float)(d & ~1);
  float den = __powf(10000.0f, i2 / (float)D_MODEL);
  float ang = (float)l / den;
  return (d & 1) ? __cosf(ang) : __sinf(ang);
}

// ---------------- embedding + posenc ----------------
__global__ void k_embed(const int* __restrict__ x, const float* __restrict__ emb,
                        float* __restrict__ h32, bf16* __restrict__ hbf) {
  int bl = blockIdx.x;              // 0..BS*L_PH-1
  int d  = threadIdx.x;             // 0..383
  int l  = bl & (L_PH - 1);
  int tok = x[bl];
  float v = emb[(long)tok * D_MODEL + d] + posenc(l, d);
  long o = (long)bl * D_MODEL + d;
  h32[o] = v;
  hbf[o] = (bf16)v;
}

// ---------------- WMMA GEMM: C = im2col(A) @ Wt^T + bias ----------------
// A: bf16 activations, rows = B*S (S = 1<<sShift), row length Cin.
// taps=1 plain / taps=3 conv-SAME (im2col on the fly; Cin % 32 == 0).
// Wt: bf16, (N, Keff) row-major (pre-transposed). Keff % 64 == 0.
// Double-buffered LDS, 64-K macro steps: 8 WMMAs per barrier.
// mode: 0 = f32 out[m*N+n]; 1 = bf16 out; 2 = relu + bf16; 3 = f32 mel-transposed (B,80,T)
__global__ __launch_bounds__(128)
void k_gemm(const bf16* __restrict__ A, const bf16* __restrict__ Wt,
            const float* __restrict__ bias,
            float* __restrict__ out32, bf16* __restrict__ outbf,
            int M, int N, int Keff, int sShift, int Cin, int taps, int mode) {
  __shared__ __align__(16) bf16 Ab[2][64][72];  // 2 bufs, 64 rows x 64 k (+pad, 144B rows)
  __shared__ __align__(16) bf16 Bb[2][64][72];

  const int S = 1 << sShift;
  int tid   = threadIdx.x;
  int lane  = tid & 31;
  int wave  = tid >> 5;
  int wm    = wave & 1, wn = wave >> 1;
  int lhalf = lane >> 4, l16 = lane & 15;
  int mBase = blockIdx.y * 64;
  int nBase = blockIdx.x * 64;

  int srow  = tid >> 1;             // staging row 0..63
  int shalf = tid & 1;              // which 32-k half of the 64-k slab (32 bf16 = 64B)

  // ---- loop-invariant row mapping (no division: S is a power of two) ----
  int r = mBase + srow;
  bool rowOk = (r < M);
  int bIdx = r >> sShift;
  int tIdx = r - (bIdx << sShift);
  const long rowBase = ((long)bIdx << sShift) * Cin;     // b*S*Cin
  const int tsh = taps >> 1;                             // 1 for k=3 SAME conv

  int nRow = nBase + srow;
  bool nOk = (nRow < N);
  const bf16* wRow = Wt + (long)nRow * Keff + shalf * 32;

  // incremental tap / cin tracking for the im2col A stream (step 64, Cin >= 64)
  int tap = 0, cin0 = shalf * 32;                        // < 64 <= Cin

  uint4 pa[4], pb[4];
  auto loadA = [&]() {
    int tt = tIdx + tap - tsh;
    if (rowOk && (tt >= 0) && (tt < S)) {
      const uint4* s = (const uint4*)(A + rowBase + (long)tt * Cin + cin0);
      pa[0] = s[0]; pa[1] = s[1]; pa[2] = s[2]; pa[3] = s[3];
    } else {
      pa[0] = pa[1] = pa[2] = pa[3] = make_uint4(0u, 0u, 0u, 0u);
    }
  };
  auto loadB = [&](int k0) {
    if (nOk) {
      const uint4* s = (const uint4*)(wRow + k0);
      pb[0] = s[0]; pb[1] = s[1]; pb[2] = s[2]; pb[3] = s[3];
    } else {
      pb[0] = pb[1] = pb[2] = pb[3] = make_uint4(0u, 0u, 0u, 0u);
    }
  };
  auto store = [&](int buf) {
    uint4* dA = (uint4*)&Ab[buf][srow][shalf * 32];
    dA[0] = pa[0]; dA[1] = pa[1]; dA[2] = pa[2]; dA[3] = pa[3];
    uint4* dB = (uint4*)&Bb[buf][srow][shalf * 32];
    dB[0] = pb[0]; dB[1] = pb[1]; dB[2] = pb[2]; dB[3] = pb[3];
  };

  v8f c[2][2];
  c[0][0] = (v8f){}; c[0][1] = (v8f){}; c[1][0] = (v8f){}; c[1][1] = (v8f){};

  const int nIter = Keff >> 6;
  // prologue: tile 0 -> buf 0
  loadA();
  loadB(0);
  store(0);
  __syncthreads();

  for (int it = 0; it < nIter; ++it) {
    int cur = it & 1;
    bool more = (it + 1) < nIter;
    if (more) {                      // issue next tile's global loads; overlap WMMAs below
      cin0 += 64;
      if (cin0 >= Cin) { cin0 -= Cin; ++tap; }
      loadA();
      loadB((it + 1) << 6);
    }
#pragma unroll
    for (int sub = 0; sub < 2; ++sub) {
      v16bf a[2], b[2];
#pragma unroll
      for (int j = 0; j < 16; ++j) {
        // A 16x32 bf16 layout: k = 16*(j>>3) + 8*lhalf + (j&7)
        int ka = sub * 32 + ((j >> 3) << 4) + (lhalf << 3) + (j & 7);
        // B 32x16 bf16 layout: k = 16*(j>>3) + 8*lhalf + (j&7)
        int kb = sub * 32 + ((j >> 3) << 4) + (lhalf << 3) + (j & 7);
        a[0][j] = Ab[cur][wm * 32 +      l16][ka];
        a[1][j] = Ab[cur][wm * 32 + 16 + l16][ka];
        b[0][j] = Bb[cur][wn * 32 +      l16][kb];
        b[1][j] = Bb[cur][wn * 32 + 16 + l16][kb];
      }
#pragma unroll
      for (int ti = 0; ti < 2; ++ti)
#pragma unroll
        for (int tj = 0; tj < 2; ++tj)
          c[ti][tj] = __builtin_amdgcn_wmma_f32_16x16x32_bf16(
              false, a[ti], false, b[tj], (short)0, c[ti][tj], false, false);
    }
    if (more) {
      store(cur ^ 1);                // publish next tile; single barrier per macro-step
      __syncthreads();
    }
  }

  // epilogue: bias (+relu) and store. C layout: m = i + 8*lhalf, n = l16.
#pragma unroll
  for (int ti = 0; ti < 2; ++ti)
#pragma unroll
    for (int tj = 0; tj < 2; ++tj) {
      int n = nBase + wn * 32 + tj * 16 + l16;
#pragma unroll
      for (int i = 0; i < 8; ++i) {
        int m = mBase + wm * 32 + ti * 16 + i + 8 * lhalf;
        if (m < M && n < N) {
          float v = c[ti][tj][i] + bias[n];
          if (mode == 2) v = fmaxf(v, 0.0f);
          if (mode == 0) {
            out32[(long)m * N + n] = v;
          } else if (mode == 3) {
            int b = m >> 9, t = m & (T_MEL - 1);
            out32[((long)b * N_MELS + n) * T_MEL + t] = v;
          } else {
            outbf[(long)m * N + n] = (bf16)v;
          }
        }
      }
    }
}

// ---------------- attention: one wave32 per (b, head, query row), online softmax ----------------
__global__ __launch_bounds__(128)
void k_attn(const bf16* __restrict__ q, const bf16* __restrict__ k,
            const bf16* __restrict__ v, bf16* __restrict__ o, int sShift) {
  const int S = 1 << sShift;
  int lane = threadIdx.x & 31;
  int gw = blockIdx.x * 4 + (threadIdx.x >> 5);
  int total = BS_ * N_HEADS * S;
  if (gw >= total) return;
  int sq = gw & (S - 1);
  int bh = gw >> sShift;
  int h = bh & 1, b = bh >> 1;
  long rowq = ((long)b * S + sq) * D_MODEL + h * DH;
  float qr[6];
#pragma unroll
  for (int j = 0; j < 6; ++j) qr[j] = (float)q[rowq + lane + 32 * j];
  const float scale = 0.07216878364870323f;   // 1/sqrt(192)
  float acc[6] = {0.f, 0.f, 0.f, 0.f, 0.f, 0.f};
  float mx = -3.0e38f, den = 0.0f;
  for (int t = 0; t < S; ++t) {
    long rk = ((long)b * S + t) * D_MODEL + h * DH;
    float p = 0.0f;
#pragma unroll
    for (int j = 0; j < 6; ++j) p += qr[j] * (float)k[rk + lane + 32 * j];
#pragma unroll
    for (int off = 16; off >= 1; off >>= 1) p += __shfl_xor(p, off, 32);
    float s = p * scale;
    float mn = fmaxf(mx, s);
    float ex = __expf(s - mn);
    float co = __expf(mx - mn);
    den = den * co + ex;
#pragma unroll
    for (int j = 0; j < 6; ++j) acc[j] = acc[j] * co + ex * (float)v[rk + lane + 32 * j];
    mx = mn;
  }
  float inv = 1.0f / den;
#pragma unroll
  for (int j = 0; j < 6; ++j) o[rowq + lane + 32 * j] = (bf16)(acc[j] * inv);
}

// ---------------- residual add + LayerNorm (one wave per row of 384) ----------------
__global__ __launch_bounds__(128)
void k_addln(const float* h, const float* __restrict__ dlt,
             const float* __restrict__ g, const float* __restrict__ bta,
             float* hout, bf16* __restrict__ hbf, int M) {
  int lane = threadIdx.x & 31;
  int row = blockIdx.x * 4 + (threadIdx.x >> 5);
  if (row >= M) return;
  long base = (long)row * D_MODEL;
  float x[12];
  float s = 0.f;
#pragma unroll
  for (int j = 0; j < 12; ++j) {
    x[j] = h[base + lane + 32 * j] + dlt[base + lane + 32 * j];
    s += x[j];
  }
#pragma unroll
  for (int off = 16; off >= 1; off >>= 1) s += __shfl_xor(s, off, 32);
  float mu = s * (1.0f / 384.0f);
  float vs = 0.f;
#pragma unroll
  for (int j = 0; j < 12; ++j) { float t = x[j] - mu; vs += t * t; }
#pragma unroll
  for (int off = 16; off >= 1; off >>= 1) vs += __shfl_xor(vs, off, 32);
  float rstd = rsqrtf(vs * (1.0f / 384.0f) + 1e-5f);
#pragma unroll
  for (int j = 0; j < 12; ++j) {
    int dd = lane + 32 * j;
    float y = (x[j] - mu) * rstd * g[dd] + bta[dd];
    hout[base + dd] = y;
    hbf[base + dd] = (bf16)y;
  }
}

// ---------------- duration cumsum ----------------
__global__ void k_cumsum(const int* __restrict__ y, int* __restrict__ cs) {
  int b = threadIdx.x;
  if (b >= BS_) return;
  int run = 0;
  for (int l = 0; l < L_PH; ++l) { run += y[b * L_PH + l]; cs[b * L_PH + l] = run; }
}

// ---------------- length regulate (+posenc), one block per (b,t) ----------------
__global__ __launch_bounds__(128)
void k_regulate(const float* __restrict__ hph, const int* __restrict__ cs,
                float* __restrict__ h32, bf16* __restrict__ hbf) {
  int bt = blockIdx.x;
  int b = bt >> 9, t = bt & (T_MEL - 1);
  const int* c = cs + b * L_PH;
  int lo = 0, hi = L_PH;                 // searchsorted(cs, t, 'right')
  while (lo < hi) { int mid = (lo + hi) >> 1; if (c[mid] <= t) lo = mid + 1; else hi = mid; }
  bool valid = t < c[L_PH - 1];
  int src = lo < L_PH ? lo : L_PH - 1;
  long obase = ((long)b * T_MEL + t) * D_MODEL;
  long ibase = ((long)b * L_PH + src) * D_MODEL;
  for (int dd = threadIdx.x; dd < D_MODEL; dd += blockDim.x) {
    float vv = valid ? hph[ibase + dd] : 0.0f;
    vv += posenc(t, dd);
    h32[obase + dd] = vv;
    hbf[obase + dd] = (bf16)vv;
  }
}

// ================================================================================
extern "C" void kernel_launch(void* const* d_in, const int* in_sizes, int n_in,
                              void* d_out, int out_size, void* d_ws, size_t ws_size,
                              hipStream_t stream) {
  (void)in_sizes; (void)n_in; (void)out_size; (void)ws_size;
  auto F = [&](int i) { return (const float*)d_in[i]; };

  // ---- workspace layout ----
  char* ws = (char*)d_ws;
  size_t off = 0;
  auto alloc = [&](size_t bytes) -> char* {
    char* p = ws + off;
    off = (off + bytes + 255) & ~(size_t)255;
    return p;
  };
  const size_t szAttn = (size_t)384 * 384, szC1 = (size_t)1152 * 1536, szC2 = (size_t)4608 * 384;
  bf16* wt[12][6];
  for (int blk = 0; blk < 12; ++blk) {
    for (int j = 0; j < 4; ++j) wt[blk][j] = (bf16*)alloc(szAttn * 2);
    wt[blk][4] = (bf16*)alloc(szC1 * 2);
    wt[blk][5] = (bf16*)alloc(szC2 * 2);
  }
  bf16*  projw  = (bf16*)alloc((size_t)384 * 80 * 2);
  float* hph32  = (float*)alloc((size_t)2048 * 384 * 4);
  bf16*  hphbf  = (bf16*)alloc((size_t)2048 * 384 * 2);
  float* hmel32 = (float*)alloc((size_t)8192 * 384 * 4);
  bf16*  hmelbf = (bf16*)alloc((size_t)8192 * 384 * 2);
  bf16*  qb = (bf16*)alloc((size_t)8192 * 384 * 2);
  bf16*  kb = (bf16*)alloc((size_t)8192 * 384 * 2);
  bf16*  vb = (bf16*)alloc((size_t)8192 * 384 * 2);
  bf16*  ob = (bf16*)alloc((size_t)8192 * 384 * 2);
  float* t32  = (float*)alloc((size_t)8192 * 384 * 4);
  bf16*  ffbf = (bf16*)alloc((size_t)8192 * 1536 * 2);
  int*   cs   = (int*)alloc((size_t)2048 * 4);

  auto castT = [&](const float* in, bf16* out, int K, int N) {
    dim3 g((N + 255) / 256, K);
    k_castT<<<g, 256, 0, stream>>>(in, out, K, N);
  };

  // ---- one-time (per launch) bf16 weight conversion, transposed to (N,K) ----
  for (int blk = 0; blk < 12; ++blk) {
    int base = (blk < 6) ? (3 + blk * 16) : (99 + (blk - 6) * 16);
    castT(F(base + 0),  wt[blk][0], 384, 384);     // wq
    castT(F(base + 2),  wt[blk][1], 384, 384);     // wk
    castT(F(base + 4),  wt[blk][2], 384, 384);     // wv
    castT(F(base + 6),  wt[blk][3], 384, 384);     // wo
    castT(F(base + 10), wt[blk][4], 1152, 1536);   // conv1 (3*384, 1536)
    castT(F(base + 12), wt[blk][5], 4608, 384);    // conv2 (3*1536, 384)
  }
  castT(F(205), projw, 384, 80);

  const int* xin = (const int*)d_in[0];
  const int* yin = (const int*)d_in[1];
  k_embed<<<BS_ * L_PH, D_MODEL, 0, stream>>>(xin, F(2), hph32, hphbf);

  auto gemm = [&](const bf16* A, const bf16* W, const float* bias, float* o32, bf16* obf,
                  int M, int N, int K, int sShift, int Cin, int taps, int mode) {
    dim3 g((N + 63) / 64, (M + 63) / 64);
    k_gemm<<<g, 128, 0, stream>>>(A, W, bias, o32, obf, M, N, K, sShift, Cin, taps, mode);
  };

  auto run_stage = [&](bool mel) {
    int sShift = mel ? 9 : 7;            // S = 512 or 128 (powers of two)
    int S = 1 << sShift;
    int M = BS_ * S;
    float* h32 = mel ? hmel32 : hph32;
    bf16*  hbf = mel ? hmelbf : hphbf;
    for (int i = 0; i < 6; ++i) {
      int blk  = mel ? (6 + i) : i;
      int base = mel ? (99 + i * 16) : (3 + i * 16);
      gemm(hbf, wt[blk][0], F(base + 1), nullptr, qb, M, 384, 384, sShift, 384, 1, 1);
      gemm(hbf, wt[blk][1], F(base + 3), nullptr, kb, M, 384, 384, sShift, 384, 1, 1);
      gemm(hbf, wt[blk][2], F(base + 5), nullptr, vb, M, 384, 384, sShift, 384, 1, 1);
      int tw = BS_ * N_HEADS * S;
      k_attn<<<(tw + 3) / 4, 128, 0, stream>>>(qb, kb, vb, ob, sShift);
      gemm(ob, wt[blk][3], F(base + 7), t32, nullptr, M, 384, 384, sShift, 384, 1, 0);
      k_addln<<<(M + 3) / 4, 128, 0, stream>>>(h32, t32, F(base + 8), F(base + 9), h32, hbf, M);
      gemm(hbf, wt[blk][4], F(base + 11), nullptr, ffbf, M, 1536, 1152, sShift, 384, 3, 2);  // conv1+relu
      gemm(ffbf, wt[blk][5], F(base + 13), t32, nullptr, M, 384, 4608, sShift, 1536, 3, 0);  // conv2
      k_addln<<<(M + 3) / 4, 128, 0, stream>>>(h32, t32, F(base + 14), F(base + 15), h32, hbf, M);
    }
  };

  run_stage(false);                                   // phoneme encoder (S=128)
  k_cumsum<<<1, 32, 0, stream>>>(yin, cs);
  k_regulate<<<BS_ * T_MEL, 128, 0, stream>>>(hph32, cs, hmel32, hmelbf);
  run_stage(true);                                    // mel decoder (S=512)
  // final projection 384->80, write transposed (B, n_mels, T) f32 into d_out
  gemm(hmelbf, projw, F(206), (float*)d_out, nullptr, BS_ * T_MEL, N_MELS, 384, 9, 384, 1, 3);
}
